// MultiHeadAttention_10307921510971
// MI455X (gfx1250) — matrix operand order repaired and checked by reading, not compiled, not run on hardware
//
#include <hip/hip_runtime.h>

// ---------------------------------------------------------------------------
// MI455X / gfx1250 multi-head attention, bf16 WMMA (16x16x32) + fp32 accum.
// ---------------------------------------------------------------------------

typedef __attribute__((ext_vector_type(16))) __bf16 v16bf;
typedef __attribute__((ext_vector_type(8)))  __bf16 v8bf;
typedef __attribute__((ext_vector_type(4)))  __bf16 v4bf;
typedef __attribute__((ext_vector_type(8)))  float  v8f;

#define DEV static __device__ __forceinline__

#ifndef __has_builtin
#define __has_builtin(x) 0
#endif
#if __has_builtin(__builtin_amdgcn_global_load_async_to_lds_b128) && \
    __has_builtin(__builtin_amdgcn_s_wait_asynccnt)
#define HAVE_ASYNC_LDS 1
#else
#define HAVE_ASYNC_LDS 0
#endif

// Raw v_exp_f32 (no denormal-range ldexp/cndmask fixups). Inputs are <= 0 and
// flushing tiny probabilities to zero is exactly the desired behavior.
#if __has_builtin(__builtin_amdgcn_exp2f)
DEV float fast_exp2(float x) { return __builtin_amdgcn_exp2f(x); }
#else
DEV float fast_exp2(float x) { return exp2f(x); }
#endif
#if __has_builtin(__builtin_amdgcn_rcpf)
DEV float fast_rcp(float x) { return __builtin_amdgcn_rcpf(x); }
#else
DEV float fast_rcp(float x) { return 1.f / x; }
#endif

constexpr int Bsz = 2;
constexpr int Sq  = 2048;
constexpr int Dm  = 1024;
constexpr int Hn  = 16;
constexpr int DKc = 64;
constexpr int Mrows = Bsz * Sq;   // 4096

constexpr int KC    = 64;         // keys streamed per attention chunk
constexpr int VPAD  = 72;         // LDS row stride (elements): 144B = 36 banks,
                                  // gcd(36,64)=4 -> 16 row-lanes conflict-free

// Compose a 16-element bf16 WMMA operand from two contiguous 16-byte chunks.
DEV v16bf load_ab(const __bf16* p0, const __bf16* p1) {
  v16bf r;
  ((v8bf*)&r)[0] = *(const v8bf*)p0;
  ((v8bf*)&r)[1] = *(const v8bf*)p1;
  return r;
}

DEV v8f wmma_bf16(v16bf a, v16bf b, v8f c) {
  return __builtin_amdgcn_wmma_f32_16x16x32_bf16(
      /*neg_a=*/false, a, /*neg_b=*/false, b,
      /*c_mod=*/(short)0, c, /*reuse_a=*/false, /*reuse_b=*/false);
}

// xor-butterfly reductions across the 16 lanes of a half-wave (masks 1,2,4,8
// never flip lane bit 4, so lanes 0-15 and 16-31 reduce independently).
DEV float red_max16(float v) {
  v = fmaxf(v, __int_as_float(__builtin_amdgcn_ds_swizzle(__float_as_int(v), 0x041f)));
  v = fmaxf(v, __int_as_float(__builtin_amdgcn_ds_swizzle(__float_as_int(v), 0x081f)));
  v = fmaxf(v, __int_as_float(__builtin_amdgcn_ds_swizzle(__float_as_int(v), 0x101f)));
  v = fmaxf(v, __int_as_float(__builtin_amdgcn_ds_swizzle(__float_as_int(v), 0x201f)));
  return v;
}
DEV float red_sum16(float v) {
  v += __int_as_float(__builtin_amdgcn_ds_swizzle(__float_as_int(v), 0x041f));
  v += __int_as_float(__builtin_amdgcn_ds_swizzle(__float_as_int(v), 0x081f));
  v += __int_as_float(__builtin_amdgcn_ds_swizzle(__float_as_int(v), 0x101f));
  v += __int_as_float(__builtin_amdgcn_ds_swizzle(__float_as_int(v), 0x201f));
  return v;
}

// ---------------------------------------------------------------------------
// fp32 -> bf16 conversion (vectorized x4)
// ---------------------------------------------------------------------------
__global__ void cvt_f32_bf16(const float* __restrict__ s,
                             __bf16* __restrict__ d, int n4) {
  int i = blockIdx.x * blockDim.x + threadIdx.x;
  if (i < n4) {
    float4 f = ((const float4*)s)[i];
    v4bf o;
    o[0] = (__bf16)f.x; o[1] = (__bf16)f.y;
    o[2] = (__bf16)f.z; o[3] = (__bf16)f.w;
    ((v4bf*)d)[i] = o;
  }
}

// ---------------------------------------------------------------------------
// Y = X (M x 1024) * W^T (W stored [e,d] row-major) + bias
// One wave computes a 32x64 output tile: 2 row-tiles x 4 col-tiles,
// 8 fp32 accumulators, K loop in steps of 32 (bf16 WMMA K-depth).
//   MODE 0: bf16 out, head-split  [B,H,S,DK]
//   MODE 1: bf16 out, head-split transposed [B,H,DK,S]   (for V)
//   MODE 2: fp32 out, contiguous [M,1024] (+bias)        (final proj)
// ---------------------------------------------------------------------------
template <int MODE>
__global__ __launch_bounds__(32)
void gemm_xwt(const __bf16* __restrict__ X, const __bf16* __restrict__ W,
              const float* __restrict__ bias, void* __restrict__ outp) {
  const int lane  = threadIdx.x;
  const int ln    = lane & 15;
  const int half  = lane >> 4;
  const int abase = half * 16;
  const int m0 = blockIdx.x * 32;
  const int n0 = blockIdx.y * 64;

  v8f acc[2][4];
#pragma unroll
  for (int i = 0; i < 2; ++i)
#pragma unroll
    for (int j = 0; j < 4; ++j) acc[i][j] = (v8f){};

  const __bf16* xr0 = X + (size_t)(m0 + ln) * Dm;
  const __bf16* xr1 = X + (size_t)(m0 + 16 + ln) * Dm;

  for (int kk = 0; kk < Dm; kk += 32) {
    if (kk + 32 < Dm) {  // gfx1250 global_prefetch_b8 of next A tile
      __builtin_prefetch(xr0 + kk + 32 + abase, 0, 3);
      __builtin_prefetch(xr1 + kk + 32 + abase, 0, 3);
    }
    v16bf a0 = load_ab(xr0 + kk + abase, xr0 + kk + abase + 8);
    v16bf a1 = load_ab(xr1 + kk + abase, xr1 + kk + abase + 8);
#pragma unroll
    for (int j = 0; j < 4; ++j) {
      const __bf16* wr = W + (size_t)(n0 + j * 16 + ln) * Dm + kk + half * 16;
      v16bf b = load_ab(wr, wr + 8);
      acc[0][j] = wmma_bf16(a0, b, acc[0][j]);
      acc[1][j] = wmma_bf16(a1, b, acc[1][j]);
    }
  }

  float bval[4];
#pragma unroll
  for (int j = 0; j < 4; ++j) bval[j] = bias[n0 + j * 16 + ln];

#pragma unroll
  for (int i = 0; i < 2; ++i) {
#pragma unroll
    for (int j = 0; j < 4; ++j) {
#pragma unroll
      for (int r = 0; r < 8; ++r) {
        const int row = m0 + i * 16 + r + half * 8;   // C layout: M = r + 8*half
        const int col = n0 + j * 16 + ln;             // C layout: N = lane%16
        const float val = acc[i][j][r] + bval[j];
        if (MODE == 2) {
          ((float*)outp)[(size_t)row * Dm + col] = val;
        } else {
          const int b  = row >> 11, s = row & (Sq - 1);
          const int h  = col >> 6,  dk = col & 63;
          __bf16* o = (__bf16*)outp;
          if (MODE == 0)
            o[(((size_t)(b * Hn + h)) * Sq + s) * DKc + dk] = (__bf16)val;
          else
            o[(((size_t)(b * Hn + h)) * DKc + dk) * Sq + s] = (__bf16)val;
        }
      }
    }
  }
}

// ---------------------------------------------------------------------------
// Async staging of a 64(dk) x 64(keys) bf16 V tile into padded LDS.
// Each async b128 moves 512B (32 lanes x 16B) = 4 dk-rows.
// Builtin signature (from compiler diagnostic): arg0 = int4* in AS(1) global,
// arg1 = LDS dest, then imm offset + cpol.
// ---------------------------------------------------------------------------
#if HAVE_ASYNC_LDS
typedef int v4i_vs __attribute__((vector_size(4 * sizeof(int))));
typedef __attribute__((address_space(1))) v4i_vs gbl_v4i;
typedef __attribute__((address_space(3))) v4i_vs lds_v4i;

DEV void stage_v(const __bf16* Vb, int k0, __bf16* dst, int lane) {
  const int subrow = lane >> 3;   // 0..3
  const int c16    = lane & 7;    // 16-byte column
#pragma unroll
  for (int i = 0; i < 16; ++i) {
    const int row = i * 4 + subrow;                       // dk row
    const __bf16* g = Vb + (size_t)row * Sq + k0 + c16 * 8;
    __bf16* l = dst + row * VPAD + c16 * 8;
    __builtin_amdgcn_global_load_async_to_lds_b128(
        (gbl_v4i*)g, (lds_v4i*)l, 0, 0);
  }
}
#endif

// ---------------------------------------------------------------------------
// Flash-style attention: one wave per 16-query tile of one (b,h), streaming
// 64 keys per chunk with online softmax (log2 domain, native v_exp_f32).
//   Qh,Kh: bf16 [B,H,S,64] ; Vt: bf16 [B,H,64,S] ; Ctx: bf16 [B,S,1024]
// Per chunk: 8 score WMMAs, softmax in C-layout, P restaged via padded LDS
// into A-layout, 8 ctx WMMAs. V tile double-buffered via async LDS loads
// (overlapped with score WMMAs + softmax) when the toolchain exposes them.
// ---------------------------------------------------------------------------
__global__ __launch_bounds__(32)
void attn_kernel(const __bf16* __restrict__ Qh, const __bf16* __restrict__ Kh,
                 const __bf16* __restrict__ Vt, __bf16* __restrict__ Ctx) {
  __shared__ __bf16 P[16 * VPAD];           // P tile [16 q rows][64 keys]+pad
#if HAVE_ASYNC_LDS
  __shared__ __bf16 Vlds[2][DKc * VPAD];    // double-buffered V tile
#endif

  const int lane  = threadIdx.x;
  const int ln    = lane & 15;
  const int half  = lane >> 4;
  const int abase = half * 16;

  const int tile = blockIdx.x;
  const int qt = tile & (Sq / 16 - 1);  // query tile within sequence
  const int bh = tile >> 7;             // b*H + h
  const int q0 = qt * 16;

  const __bf16* Qb = Qh + (size_t)bh * Sq * DKc;
  const __bf16* Kb = Kh + (size_t)bh * Sq * DKc;
  const __bf16* Vb = Vt + (size_t)bh * DKc * Sq;

  // Q tile as two A operands (dk 0..31 and 32..63); constant over the K loop.
  const __bf16* qrow = Qb + (size_t)(q0 + ln) * DKc;
  const v16bf aq0 = load_ab(qrow + abase,      qrow + abase + 8);
  const v16bf aq1 = load_ab(qrow + 32 + abase, qrow + 40 + abase);

  v8f acc0 = (v8f){}, acc1 = (v8f){}, acc2 = (v8f){}, acc3 = (v8f){};
  float mr[8], lr[8];
#pragma unroll
  for (int r = 0; r < 8; ++r) { mr[r] = -1e30f; lr[r] = 0.f; }

  // softmax in log2 domain: exp(s*scale) == exp2(s*scale*log2(e))
  const float scale2 = 0.125f * 1.44269504088896340736f;

#if HAVE_ASYNC_LDS
  stage_v(Vb, 0, Vlds[0], lane);
  int buf = 0;
#endif

  for (int k0 = 0; k0 < Sq; k0 += KC) {
#if HAVE_ASYNC_LDS
    if (k0 + KC < Sq) stage_v(Vb, k0 + KC, Vlds[buf ^ 1], lane);
#endif

    // ---- scores: four 16x16 C tiles (keys k0+16t .. +15) ----
    v8f c[4];
#pragma unroll
    for (int t = 0; t < 4; ++t) {
      const __bf16* kr = Kb + (size_t)(k0 + t * 16 + ln) * DKc + half * 16;
      v16bf blo = load_ab(kr,      kr + 8);   // dk 0..31 slice of K rows
      v16bf bhi = load_ab(kr + 32, kr + 40);  // dk 32..63 slice
      v8f cc = (v8f){};
      cc = wmma_bf16(aq0, blo, cc);
      cc = wmma_bf16(aq1, bhi, cc);
      c[t] = cc;
    }

    // ---- online softmax in C layout (row = r + 8*half, col = ln) ----
#pragma unroll
    for (int r = 0; r < 8; ++r) {
      float s0 = c[0][r] * scale2;
      float s1 = c[1][r] * scale2;
      float s2 = c[2][r] * scale2;
      float s3 = c[3][r] * scale2;
      float t  = red_max16(fmaxf(fmaxf(s0, s1), fmaxf(s2, s3)));
      float mn = fmaxf(mr[r], t);
      float al = fast_exp2(mr[r] - mn);
      float p0 = fast_exp2(s0 - mn);
      float p1 = fast_exp2(s1 - mn);
      float p2 = fast_exp2(s2 - mn);
      float p3 = fast_exp2(s3 - mn);
      float rs = red_sum16((p0 + p1) + (p2 + p3));
      lr[r] = lr[r] * al + rs;
      mr[r] = mn;
      acc0[r] *= al; acc1[r] *= al; acc2[r] *= al; acc3[r] *= al;
      const int row = r + half * 8;
      P[row * VPAD + ln]      = (__bf16)p0;
      P[row * VPAD + 16 + ln] = (__bf16)p1;
      P[row * VPAD + 32 + ln] = (__bf16)p2;
      P[row * VPAD + 48 + ln] = (__bf16)p3;
    }
    __syncthreads();   // single-wave workgroup: cheap; orders LDS write->read

#if HAVE_ASYNC_LDS
    // Current buffer's 16 async loads are the oldest outstanding; the next
    // buffer's 16 (just issued) may remain in flight. Async loads retire
    // in order, so waiting to <=16 guarantees the current tile landed.
    if (k0 + KC < Sq) __builtin_amdgcn_s_wait_asynccnt(16);
    else              __builtin_amdgcn_s_wait_asynccnt(0);
#endif

    // ---- ctx += P(16x64) * V(64x64) : two K-steps x four dk blocks ----
#pragma unroll
    for (int ks = 0; ks < 2; ++ks) {
      const int s = ks * 32;
      const __bf16* pr = &P[ln * VPAD + s + abase];
      v16bf ap = load_ab(pr, pr + 8);
#if HAVE_ASYNC_LDS
      const __bf16* v0 = (const __bf16*)&Vlds[buf][(0 * 16 + ln) * VPAD + s + half * 16];
      const __bf16* v1 = (const __bf16*)&Vlds[buf][(1 * 16 + ln) * VPAD + s + half * 16];
      const __bf16* v2 = (const __bf16*)&Vlds[buf][(2 * 16 + ln) * VPAD + s + half * 16];
      const __bf16* v3 = (const __bf16*)&Vlds[buf][(3 * 16 + ln) * VPAD + s + half * 16];
#else
      const __bf16* v0 = Vb + (size_t)(0 * 16 + ln) * Sq + k0 + s + half * 16;
      const __bf16* v1 = Vb + (size_t)(1 * 16 + ln) * Sq + k0 + s + half * 16;
      const __bf16* v2 = Vb + (size_t)(2 * 16 + ln) * Sq + k0 + s + half * 16;
      const __bf16* v3 = Vb + (size_t)(3 * 16 + ln) * Sq + k0 + s + half * 16;
#endif
      acc0 = wmma_bf16(ap, load_ab(v0, v0 + 8), acc0);
      acc1 = wmma_bf16(ap, load_ab(v1, v1 + 8), acc1);
      acc2 = wmma_bf16(ap, load_ab(v2, v2 + 8), acc2);
      acc3 = wmma_bf16(ap, load_ab(v3, v3 + 8), acc3);
    }
    __syncthreads();   // protect P (and V buffer swap) before next chunk
#if HAVE_ASYNC_LDS
    buf ^= 1;
#endif
  }

  // ---- epilogue: ctx / l, store bf16 into [B,S,H*64] for final GEMM ----
  const int b = bh >> 4, h = bh & 15;
#pragma unroll
  for (int r = 0; r < 8; ++r) {
    const int row = q0 + r + half * 8;
    const float inv = fast_rcp(lr[r]);
    const size_t base = ((size_t)b * Sq + row) * Dm + h * DKc + ln;
    Ctx[base + 0]  = (__bf16)(acc0[r] * inv);
    Ctx[base + 16] = (__bf16)(acc1[r] * inv);
    Ctx[base + 32] = (__bf16)(acc2[r] * inv);
    Ctx[base + 48] = (__bf16)(acc3[r] * inv);
  }
}

// ---------------------------------------------------------------------------
// Host orchestration
// ---------------------------------------------------------------------------
extern "C" void kernel_launch(void* const* d_in, const int* in_sizes, int n_in,
                              void* d_out, int out_size, void* d_ws, size_t ws_size,
                              hipStream_t stream) {
  const float* q  = (const float*)d_in[0];
  const float* k  = (const float*)d_in[1];
  const float* v  = (const float*)d_in[2];
  const float* wq = (const float*)d_in[3];
  const float* bq = (const float*)d_in[4];
  const float* wk = (const float*)d_in[5];
  const float* bk = (const float*)d_in[6];
  const float* wv = (const float*)d_in[7];
  const float* bv = (const float*)d_in[8];
  const float* wo = (const float*)d_in[9];
  const float* bo = (const float*)d_in[10];

  char* wsp = (char*)d_ws;
  auto carve = [&](size_t bytes) -> void* {
    void* p = (void*)wsp;
    wsp += (bytes + 255) & ~(size_t)255;
    return p;
  };
  const size_t xbytes = (size_t)Mrows * Dm * sizeof(__bf16);  // 8 MB
  const size_t wbytes = (size_t)Dm * Dm * sizeof(__bf16);     // 2 MB

  __bf16* xq  = (__bf16*)carve(xbytes);
  __bf16* xk  = (__bf16*)carve(xbytes);
  __bf16* xv  = (__bf16*)carve(xbytes);
  __bf16* wqb = (__bf16*)carve(wbytes);
  __bf16* wkb = (__bf16*)carve(wbytes);
  __bf16* wvb = (__bf16*)carve(wbytes);
  __bf16* wob = (__bf16*)carve(wbytes);
  __bf16* qh  = (__bf16*)carve(xbytes);   // [B,H,S,64]
  __bf16* kh  = (__bf16*)carve(xbytes);   // [B,H,S,64]
  __bf16* vt  = (__bf16*)carve(xbytes);   // [B,H,64,S]
  __bf16* ctx = (__bf16*)carve(xbytes);   // [B,S,1024]

  const int n4x = Mrows * Dm / 4;   // 1,048,576
  const int n4w = Dm * Dm / 4;      //   262,144
  cvt_f32_bf16<<<n4x / 256, 256, 0, stream>>>(q,  xq,  n4x);
  cvt_f32_bf16<<<n4x / 256, 256, 0, stream>>>(k,  xk,  n4x);
  cvt_f32_bf16<<<n4x / 256, 256, 0, stream>>>(v,  xv,  n4x);
  cvt_f32_bf16<<<n4w / 256, 256, 0, stream>>>(wq, wqb, n4w);
  cvt_f32_bf16<<<n4w / 256, 256, 0, stream>>>(wk, wkb, n4w);
  cvt_f32_bf16<<<n4w / 256, 256, 0, stream>>>(wv, wvb, n4w);
  cvt_f32_bf16<<<n4w / 256, 256, 0, stream>>>(wo, wob, n4w);

  dim3 gg(Mrows / 32, Dm / 64);   // 128 x 16 waves
  gemm_xwt<0><<<gg, 32, 0, stream>>>(xq, wqb, bq, (void*)qh);
  gemm_xwt<0><<<gg, 32, 0, stream>>>(xk, wkb, bk, (void*)kh);
  gemm_xwt<1><<<gg, 32, 0, stream>>>(xv, wvb, bv, (void*)vt);

  attn_kernel<<<Bsz * Hn * (Sq / 16), 32, 0, stream>>>(qh, kh, vt, ctx);

  gemm_xwt<2><<<gg, 32, 0, stream>>>(ctx, wob, bo, d_out);
}
